// OAIAttention_17729624998176
// MI455X (gfx1250) — matrix operand order repaired and checked by reading, not compiled, not run on hardware
//
#include <hip/hip_runtime.h>

typedef unsigned short u16;
typedef unsigned int   u32;

typedef __bf16 v16bf __attribute__((ext_vector_type(16)));
typedef float  v8f   __attribute__((ext_vector_type(8)));
typedef int    v4i   __attribute__((ext_vector_type(4)));

union FragB16 { v16bf v; uint4 u[2]; };

#ifndef __has_builtin
#define __has_builtin(x) 0
#endif
#if __has_builtin(__builtin_amdgcn_global_load_async_to_lds_b128) && \
    __has_builtin(__builtin_amdgcn_s_wait_asynccnt)
#define USE_ASYNC_LDS 1
#else
#define USE_ASYNC_LDS 0
#endif

// ---- problem constants ----
constexpr int T     = 1024;
constexpr int T_PAD = 1024 + 32;               // padded key axis for V^T
constexpr int HID   = 2880;
constexpr int H     = 64;
constexpr int KVH   = 8;
constexpr int D     = 64;
constexpr int QKV_OUT = H * D + 2 * KVH * D;   // 5120
constexpr int HD    = H * D;                   // 4096
constexpr int KVD   = KVH * D;                 // 512

__device__ inline u16 f2bf(float f) {
  union { float f; u32 u; } x; x.f = f;
  u32 r = x.u + 0x7FFFu + ((x.u >> 16) & 1u);
  return (u16)(r >> 16);
}

__device__ inline v8f wmma_bf16(v16bf a, v16bf b, v8f c) {
  return __builtin_amdgcn_wmma_f32_16x16x32_bf16(false, a, false, b, (short)0, c, false, false);
}

#if USE_ASYNC_LDS
// 16B async DMA global -> LDS (ASYNCcnt-tracked, no VGPR staging)
// builtin signature (from diagnostics): (v4i AS1* src, v4i AS3* dst, imm off, imm cpol)
__device__ inline void async_cp16(u16* lds, const u16* g) {
  const v4i* gp = (const v4i*)g;
  v4i* lp = (v4i*)lds;
  __builtin_amdgcn_global_load_async_to_lds_b128(
      (__attribute__((address_space(1))) v4i*)gp,
      (__attribute__((address_space(3))) v4i*)lp, 0, 0);
}
#endif

// ---------------- kernel 1: RMSNorm -> bf16 ----------------
__global__ __launch_bounds__(256) void k_rmsnorm(const float* __restrict__ x,
                                                 const float* __restrict__ g,
                                                 u16* __restrict__ o) {
  int row = blockIdx.x, tid = threadIdx.x;
  __shared__ float red[256];
  const float* xr = x + (size_t)row * HID;
  float s = 0.f;
  for (int i = tid; i < HID; i += 256) { float v = xr[i]; s += v * v; }
  red[tid] = s;
  __syncthreads();
  for (int off = 128; off > 0; off >>= 1) {
    if (tid < off) red[tid] += red[tid + off];
    __syncthreads();
  }
  float rms = rsqrtf(red[0] / (float)HID + 1e-5f);
  for (int i = tid; i < HID; i += 256)
    o[(size_t)row * HID + i] = f2bf(xr[i] * rms * g[i]);
}

// ------- kernel 2: fp32 [K][N] -> bf16 transposed [N][K] (tiled, coalesced) -------
__global__ __launch_bounds__(256) void k_cvt_t(const float* __restrict__ in,
                                               u16* __restrict__ out, int K, int N) {
  __shared__ float tile[32][33];
  int k0 = blockIdx.y * 32, n0 = blockIdx.x * 32;
  int tr = threadIdx.x >> 5, tc = threadIdx.x & 31;   // tr: 0..7
  #pragma unroll
  for (int i = 0; i < 4; ++i)
    tile[tr + 8 * i][tc] = in[(size_t)(k0 + tr + 8 * i) * N + n0 + tc];
  __syncthreads();
  #pragma unroll
  for (int i = 0; i < 4; ++i)
    out[(size_t)(n0 + tr + 8 * i) * K + k0 + tc] = f2bf(tile[tc][tr + 8 * i]);
}

// ---- kernel 3/6: WMMA GEMM  C = A(bf16 [M,K]) x Bt(bf16 [N,K]) + bias (+resid) ----
// block tile 128x128, 8 waves (2 M x 4 N), double-buffered LDS.
__global__ __launch_bounds__(256) void k_gemm_bf16(const u16* __restrict__ A,
                                                   const u16* __restrict__ Bt,
                                                   const float* __restrict__ bias,
                                                   const float* __restrict__ resid,
                                                   float* __restrict__ C,
                                                   int M, int N, int K) {
  __shared__ u16 sA[2][128 * 32];    // [row 128][k 32]
  __shared__ u16 sB[2][128 * 32];    // [col 128][k 32]
  int tid = threadIdx.x;
  int m0 = blockIdx.y * 128, n0 = blockIdx.x * 128;
  int w = tid >> 5, lane = tid & 31;
  int wM = (w & 1) * 64, wN = (w >> 1) * 32;
  int lrow = lane & 15, hi = lane >> 4;
  int k0a = hi ? 8 : 0;    // A fragment chunk base (ISA 16-bit A layout)
  int k0b = hi ? 8 : 0;    // B fragment chunk base (same k order as A)

  // staging index: element lin = (it*256+tid)*8 -> row lin>>5, k lin&31
  int r0[2], c0[2], bn[2];
  #pragma unroll
  for (int it = 0; it < 2; ++it) {
    int lin = (it * 256 + tid) << 3;
    r0[it] = lin >> 5;
    c0[it] = lin & 31;
    int gn = n0 + r0[it];
    bn[it] = gn < N ? gn : N - 1;   // clamp: garbage cols never stored (see epilogue)
  }

  v8f acc[8];
  #pragma unroll
  for (int i = 0; i < 8; ++i)
    #pragma unroll
    for (int r = 0; r < 8; ++r) acc[i][r] = 0.f;

  int ksteps = K >> 5;

  auto compute = [&](int b) {
    FragB16 fa[4], fb[2];
    #pragma unroll
    for (int mi = 0; mi < 4; ++mi) {
      const u16* p = &sA[b][(wM + mi * 16 + lrow) * 32 + k0a];
      fa[mi].u[0] = *(const uint4*)p;
      fa[mi].u[1] = *(const uint4*)(p + 16);
    }
    #pragma unroll
    for (int ni = 0; ni < 2; ++ni) {
      const u16* p = &sB[b][(wN + ni * 16 + lrow) * 32 + k0b];
      fb[ni].u[0] = *(const uint4*)p;
      fb[ni].u[1] = *(const uint4*)(p + 16);
    }
    #pragma unroll
    for (int mi = 0; mi < 4; ++mi)
      #pragma unroll
      for (int ni = 0; ni < 2; ++ni)
        acc[mi * 2 + ni] = wmma_bf16(fa[mi].v, fb[ni].v, acc[mi * 2 + ni]);
  };

#if USE_ASYNC_LDS
  // -------- async DMA pipeline: global -> LDS without VGPR staging --------
  auto issue = [&](int b, int k0) {   // exactly 4 async b128 per lane
    #pragma unroll
    for (int it = 0; it < 2; ++it)
      async_cp16(&sA[b][r0[it] * 32 + c0[it]], &A[(size_t)(m0 + r0[it]) * K + k0 + c0[it]]);
    #pragma unroll
    for (int it = 0; it < 2; ++it)
      async_cp16(&sB[b][r0[it] * 32 + c0[it]], &Bt[(size_t)bn[it] * K + k0 + c0[it]]);
  };
  issue(0, 0);
  int buf = 0;
  for (int kt = 0; kt < ksteps; ++kt) {
    if (kt + 1 < ksteps) {
      issue(buf ^ 1, (kt + 1) << 5);          // prev read of buf^1 fenced by trailing barrier
      __builtin_amdgcn_s_wait_asynccnt(4);    // in-order: buf's group has landed
    } else {
      __builtin_amdgcn_s_wait_asynccnt(0);
    }
    __syncthreads();
    compute(buf);
    __syncthreads();                           // all readers done before buf is re-filled
    buf ^= 1;
  }
#else
  // -------- fallback: register-staged double-buffer pipeline --------
  uint4 rA[2], rB[2];
  auto gload = [&](int k0) {
    #pragma unroll
    for (int it = 0; it < 2; ++it) {
      rA[it] = *(const uint4*)&A[(size_t)(m0 + r0[it]) * K + k0 + c0[it]];
      rB[it] = *(const uint4*)&Bt[(size_t)bn[it] * K + k0 + c0[it]];
    }
  };
  gload(0);
  int buf = 0;
  for (int kt = 0; kt < ksteps; ++kt) {
    #pragma unroll
    for (int it = 0; it < 2; ++it) {
      *(uint4*)&sA[buf][r0[it] * 32 + c0[it]] = rA[it];
      *(uint4*)&sB[buf][r0[it] * 32 + c0[it]] = rB[it];
    }
    __syncthreads();
    if (kt + 1 < ksteps) {
      gload((kt + 1) << 5);
      if (kt + 2 < ksteps) {
        int pk = (kt + 2) << 5;
        __builtin_prefetch(&A[(size_t)(m0 + r0[0]) * K + pk], 0, 1);
        __builtin_prefetch(&Bt[(size_t)bn[0] * K + pk], 0, 1);
      }
    }
    compute(buf);
    buf ^= 1;
  }
#endif

  // epilogue: C-layout lane holds col=lrow, rows hi*8 + r
  #pragma unroll
  for (int mi = 0; mi < 4; ++mi)
    #pragma unroll
    for (int ni = 0; ni < 2; ++ni) {
      int gn = n0 + wN + ni * 16 + lrow;
      if (gn >= N) continue;
      float bv = bias ? bias[gn] : 0.f;
      #pragma unroll
      for (int r = 0; r < 8; ++r) {
        int gm = m0 + wM + mi * 16 + hi * 8 + r;
        float v = acc[mi * 2 + ni][r] + bv;
        if (resid) v += resid[(size_t)gm * N + gn];
        C[(size_t)gm * N + gn] = v;
      }
    }
}

// ---------------- kernel 4: RoPE (YaRN) + split + bf16 ----------------
// qb: [T][H*D]   kb: [T][KVH*D]   vT: [KVH*D][T_PAD] (transposed, zero-padded tail)
__global__ __launch_bounds__(256) void k_rope_split(const float* __restrict__ qkv,
                                                    const int* __restrict__ positions,
                                                    u16* __restrict__ qb,
                                                    u16* __restrict__ kb,
                                                    u16* __restrict__ vT) {
  constexpr double kPI = 3.14159265358979323846;
  int t = blockIdx.x, tid = threadIdx.x;
  if (t >= T) {                      // zero the padded key tail of V^T
    for (int i = tid; i < KVD; i += 256) vT[(size_t)i * T_PAD + t] = 0;
    return;
  }
  __shared__ float cs[32], sn[32];
  if (tid < 32) {
    double d = (double)tid;
    double lnB = log(150000.0);
    double freq = exp(lnB * d / 32.0);                 // BASE^(d/32)
    double conc = 0.1 * log(32.0) + 1.0;
    double low  = 32.0 * log(4096.0 / (32.0 * 2.0 * kPI)) / lnB;
    double high = 32.0 * log(4096.0 / (1.0  * 2.0 * kPI)) / lnB;
    double interp = 1.0 / (32.0 * freq);
    double extrap = 1.0 / freq;
    double ramp = (d - low) / (high - low);
    double mask = 1.0 - fmin(fmax(ramp, 0.0), 1.0);
    double invf = interp * (1.0 - mask) + extrap * mask;
    double ang = (double)positions[t] * invf;
    cs[tid] = (float)(cos(ang) * conc);
    sn[tid] = (float)(sin(ang) * conc);
  }
  __syncthreads();
  const float* row = qkv + (size_t)t * QKV_OUT;
  // Q: fold in attention scale 1/sqrt(D)
  for (int i = tid; i < H * 32; i += 256) {
    int h = i >> 5, d = i & 31;
    float x1 = row[h * 64 + d], x2 = row[h * 64 + d + 32];
    qb[(size_t)t * HD + h * 64 + d]      = f2bf((x1 * cs[d] - x2 * sn[d]) * 0.125f);
    qb[(size_t)t * HD + h * 64 + d + 32] = f2bf((x2 * cs[d] + x1 * sn[d]) * 0.125f);
  }
  // K
  for (int i = tid; i < KVH * 32; i += 256) {
    int h = i >> 5, d = i & 31;
    float x1 = row[HD + h * 64 + d], x2 = row[HD + h * 64 + d + 32];
    kb[(size_t)t * KVD + h * 64 + d]      = f2bf(x1 * cs[d] - x2 * sn[d]);
    kb[(size_t)t * KVD + h * 64 + d + 32] = f2bf(x2 * cs[d] + x1 * sn[d]);
  }
  // V transposed: vT[dkv][t]
  for (int i = tid; i < KVD; i += 256)
    vT[(size_t)i * T_PAD + t] = f2bf(row[HD + KVD + i]);
}

// ---------------- kernel 5: sliding-window attention with sinks (WMMA) ------
// 1 wave per (16-query tile, head). Keys [q0-128, q0+31] in 5 chunks of 32.
__global__ __launch_bounds__(32) void k_attn(const u16* __restrict__ qb,
                                             const u16* __restrict__ kbuf,
                                             const u16* __restrict__ vT,
                                             const float* __restrict__ sinks,
                                             u16* __restrict__ attnb) {
  int qt = blockIdx.x;           // 0..63
  int h  = blockIdx.y;           // 0..63
  int q0 = qt * 16;
  int kvh = h >> 3;
  int lane = threadIdx.x;
  int lrow = lane & 15, hi = lane >> 4;
  int k0a = hi ? 8 : 0, k0b = hi ? 8 : 0;

  __shared__ u16  sQ[16 * 64];
  __shared__ u16  sK[32 * 64];     // [key][dim] row-major
  __shared__ u16  sVt[64 * 32];    // [dim][key]
  __shared__ float sS[16 * 32];
  __shared__ u16  sP[16 * 32];
  __shared__ float sAl[16];
  __shared__ float sSc[16];

  // stage Q tile (16x64)
#if USE_ASYNC_LDS
  for (int i = lane; i < 128; i += 32) {
    int r = i >> 3, c8 = (i & 7) * 8;
    async_cp16(&sQ[r * 64 + c8], &qb[(size_t)(q0 + r) * HD + h * 64 + c8]);
  }
  __builtin_amdgcn_s_wait_asynccnt(0);
#else
  for (int i = lane; i < 128; i += 32) {
    int r = i >> 3, c8 = (i & 7) * 8;
    *(uint4*)&sQ[r * 64 + c8] = *(const uint4*)&qb[(size_t)(q0 + r) * HD + h * 64 + c8];
  }
#endif
  __syncthreads();

  FragB16 fq[2];
  #pragma unroll
  for (int ks = 0; ks < 2; ++ks) {
    const u16* p = &sQ[lrow * 64 + ks * 32 + k0a];
    fq[ks].u[0] = *(const uint4*)p;
    fq[ks].u[1] = *(const uint4*)(p + 16);
  }

  v8f acc[4];
  #pragma unroll
  for (int n = 0; n < 4; ++n)
    #pragma unroll
    for (int r = 0; r < 8; ++r) acc[n][r] = 0.f;
  float m = -INFINITY, l = 0.f;    // row state lives in lanes 0..15 (row = lane)

  for (int c = 0; c < 5; ++c) {
    int kbase = q0 - 128 + c * 32;
    if (kbase < 0) continue;       // uniform skip
    // stage K chunk (32x64) row-major (clamp rows >= T; masked later)
    // and V chunk from transposed global V (contiguous b128, zero-padded tail)
#if USE_ASYNC_LDS
    for (int i = lane; i < 256; i += 32) {
      int r = i >> 3, c8 = (i & 7) * 8;
      int krow = kbase + r; if (krow > T - 1) krow = T - 1;
      async_cp16(&sK[r * 64 + c8], &kbuf[(size_t)krow * KVD + kvh * 64 + c8]);
    }
    for (int i = lane; i < 256; i += 32) {
      int d = i >> 2, kk8 = (i & 3) * 8;
      async_cp16(&sVt[d * 32 + kk8], &vT[(size_t)(kvh * 64 + d) * T_PAD + kbase + kk8]);
    }
    __builtin_amdgcn_s_wait_asynccnt(0);
#else
    for (int i = lane; i < 256; i += 32) {
      int r = i >> 3, c8 = (i & 7) * 8;
      int krow = kbase + r; if (krow > T - 1) krow = T - 1;
      *(uint4*)&sK[r * 64 + c8] =
          *(const uint4*)&kbuf[(size_t)krow * KVD + kvh * 64 + c8];
    }
    for (int i = lane; i < 256; i += 32) {
      int d = i >> 2, kk8 = (i & 3) * 8;
      *(uint4*)&sVt[d * 32 + kk8] =
          *(const uint4*)&vT[(size_t)(kvh * 64 + d) * T_PAD + kbase + kk8];
    }
#endif
    __syncthreads();

    // S = Q K^T  (two 16x16 N-tiles over 32 keys)
    #pragma unroll
    for (int n = 0; n < 2; ++n) {
      v8f s;
      #pragma unroll
      for (int r = 0; r < 8; ++r) s[r] = 0.f;
      #pragma unroll
      for (int ks = 0; ks < 2; ++ks) {
        FragB16 fk;
        const u16* p = &sK[(n * 16 + lrow) * 64 + ks * 32 + k0b];
        fk.u[0] = *(const uint4*)p;
        fk.u[1] = *(const uint4*)(p + 16);
        s = wmma_bf16(fq[ks].v, fk.v, s);
      }
      #pragma unroll
      for (int r = 0; r < 8; ++r)
        sS[(hi * 8 + r) * 32 + n * 16 + lrow] = s[r];
    }
    __syncthreads();

    // online softmax, lanes 0..15 own one row each
    if (lane < 16) {
      int qi = q0 + lane;
      float cmax = -INFINITY;
      float sv[32];
      #pragma unroll
      for (int j = 0; j < 32; ++j) {
        int kj = kbase + j;
        bool ok = (kj <= qi) && (kj > qi - 128);
        float x = ok ? sS[lane * 32 + j] : -INFINITY;
        sv[j] = x;
        cmax = fmaxf(cmax, x);
      }
      float nm = fmaxf(m, cmax);
      float alpha;
      float sum = 0.f;
      if (nm == -INFINITY) {
        alpha = 1.f;
        #pragma unroll
        for (int j = 0; j < 32; ++j) sP[lane * 32 + j] = 0;
      } else {
        alpha = __expf(m - nm);
        #pragma unroll
        for (int j = 0; j < 32; ++j) {
          float p = __expf(sv[j] - nm);
          sum += p;
          sP[lane * 32 + j] = f2bf(p);
        }
      }
      l = l * alpha + sum;
      m = nm;
      sAl[lane] = alpha;
    }
    __syncthreads();

    // rescale O, then O += P V   (P: 16x32 A-frag, V: 32x64 via 4 B-frags)
    float al[8];
    #pragma unroll
    for (int r = 0; r < 8; ++r) al[r] = sAl[hi * 8 + r];
    FragB16 fp;
    {
      const u16* p = &sP[lrow * 32 + k0a];
      fp.u[0] = *(const uint4*)p;
      fp.u[1] = *(const uint4*)(p + 16);
    }
    #pragma unroll
    for (int n = 0; n < 4; ++n) {
      #pragma unroll
      for (int r = 0; r < 8; ++r) acc[n][r] *= al[r];
      FragB16 fv;
      const u16* p = &sVt[(n * 16 + lrow) * 32 + k0b];
      fv.u[0] = *(const uint4*)p;
      fv.u[1] = *(const uint4*)(p + 16);
      acc[n] = wmma_bf16(fp.v, fv.v, acc[n]);
    }
    __syncthreads();
  }

  // final normalization including sink: 1 / (l + exp(sink - m))
  if (lane < 16) {
    float denom = l + __expf(sinks[h] - m);
    sSc[lane] = 1.f / denom;
  }
  __syncthreads();
  float sc[8];
  #pragma unroll
  for (int r = 0; r < 8; ++r) sc[r] = sSc[hi * 8 + r];
  #pragma unroll
  for (int n = 0; n < 4; ++n)
    #pragma unroll
    for (int r = 0; r < 8; ++r) {
      int row = q0 + hi * 8 + r;
      int d = n * 16 + lrow;
      attnb[(size_t)row * HD + h * 64 + d] = f2bf(acc[n][r] * sc[r]);
    }
}

// ---------------- host launcher ----------------
extern "C" void kernel_launch(void* const* d_in, const int* in_sizes, int n_in,
                              void* d_out, int out_size, void* d_ws, size_t ws_size,
                              hipStream_t stream) {
  const float* hidden     = (const float*)d_in[0];
  const int*   positions  = (const int*)d_in[1];
  const float* norm_scale = (const float*)d_in[2];
  const float* w_qkv      = (const float*)d_in[3];
  const float* b_qkv      = (const float*)d_in[4];
  const float* w_out      = (const float*)d_in[5];
  const float* b_out      = (const float*)d_in[6];
  const float* sinks      = (const float*)d_in[7];
  float* out = (float*)d_out;

  char* wsb = (char*)d_ws;
  size_t off = 0;
  auto alloc = [&](size_t bytes) -> void* {
    void* p = wsb + off;
    off = (off + bytes + 255) & ~(size_t)255;
    return p;
  };
  u16*   t_bf     = (u16*)alloc((size_t)T * HID * 2);
  u16*   wqkv_bfT = (u16*)alloc((size_t)HID * QKV_OUT * 2);   // [5120][2880]
  u16*   wout_bfT = (u16*)alloc((size_t)HD * HID * 2);        // [2880][4096]
  float* qkv_f    = (float*)alloc((size_t)T * QKV_OUT * 4);
  u16*   q_bf     = (u16*)alloc((size_t)T * HD * 2);
  u16*   k_bf     = (u16*)alloc((size_t)T * KVD * 2);
  u16*   vT_bf    = (u16*)alloc((size_t)KVD * T_PAD * 2);
  u16*   attn_bf  = (u16*)alloc((size_t)T * HD * 2);
  (void)ws_size; (void)in_sizes; (void)n_in; (void)out_size;

  // 1) RMSNorm -> bf16 activations
  k_rmsnorm<<<T, 256, 0, stream>>>(hidden, norm_scale, t_bf);
  // 2) weights -> bf16, transposed to [N][K]
  k_cvt_t<<<dim3(QKV_OUT / 32, HID / 32), 256, 0, stream>>>(w_qkv, wqkv_bfT, HID, QKV_OUT);
  k_cvt_t<<<dim3(HID / 32, HD / 32), 256, 0, stream>>>(w_out, wout_bfT, HD, HID);
  // 3) QKV GEMM (+bias): [1024,2880] x [2880,5120] -> fp32
  k_gemm_bf16<<<dim3(QKV_OUT / 128, T / 128), 256, 0, stream>>>(
      t_bf, wqkv_bfT, b_qkv, nullptr, qkv_f, T, QKV_OUT, HID);
  // 4) RoPE + split + bf16 (scale folded into Q; V stored transposed, padded)
  k_rope_split<<<T_PAD, 256, 0, stream>>>(qkv_f, positions, q_bf, k_bf, vT_bf);
  // 5) sliding-window attention with sinks
  k_attn<<<dim3(T / 16, H), 32, 0, stream>>>(q_bf, k_bf, vT_bf, sinks, attn_bf);
  // 6) out GEMM + bias + residual: [1024,4096] x [4096,2880] -> d_out fp32
  k_gemm_bf16<<<dim3((HID + 127) / 128, T / 128), 256, 0, stream>>>(
      attn_bf, wout_bfT, b_out, hidden, out, T, HID, HD);
}
